// MultiBoxHeads_24335284699235
// MI455X (gfx1250) — compile-verified, matrix operand order checked by reading, not run on hardware
//
#include <hip/hip_runtime.h>
#include <math.h>

#define AS1 __attribute__((address_space(1)))
#define AS3 __attribute__((address_space(3)))

#if __has_builtin(__builtin_amdgcn_global_load_async_to_lds_b32)
#define HAVE_ASYNC_LDS 1
#endif

typedef __attribute__((ext_vector_type(16))) _Float16 v16h;
typedef __attribute__((ext_vector_type(8)))  float    v8f;

static constexpr int Bc = 64, Pc = 8732, Cc = 81;
static constexpr int PITCH = 96;                 // 81 padded to 3x32 K-chunks
static constexpr int WPB = 4;                    // waves per block (128 threads)
static constexpr int G1X = (Pc + 16 * WPB - 1) / (16 * WPB);  // 137

__device__ __forceinline__ unsigned float_to_key(float f) {
  unsigned u = __float_as_uint(f);
  return (u & 0x80000000u) ? ~u : (u | 0x80000000u);  // larger key == larger float
}

// ---------------- Phase 1: logsumexp (WMMA), keys/ce, pos counts, smooth-L1 ----
__global__ __launch_bounds__(128) void mb_phase1(
    const float* __restrict__ conf, const float* __restrict__ ploc,
    const float* __restrict__ gloc, const int* __restrict__ labels,
    unsigned* __restrict__ keys, float* __restrict__ cearr,
    int* __restrict__ npr, float* __restrict__ sl1p)
{
  __shared__ float tile[WPB][16 * PITCH];   // 24 KB, -inf padded
  __shared__ float rowsum[WPB][16];
  __shared__ float wsl1[WPB];

  const int tid  = threadIdx.x;
  const int w    = tid >> 5;
  const int lane = tid & 31;
  const int b    = blockIdx.y;
  const int p0   = (blockIdx.x * WPB + w) * 16;
  const int nv   = min(16, Pc - p0);        // wave-uniform; may be <= 0
  const int row  = lane & 15;
  const float NEG_INF = -__builtin_inff();

  const float* gconf = conf + ((size_t)b * Pc + p0) * Cc;

  if (nv > 0) {
    const int nelem = nv * Cc;
#ifdef HAVE_ASYNC_LDS
    for (int idx = lane; idx < nelem; idx += 32) {
      int r = idx / Cc, c = idx - r * Cc;
      AS1 int* g = (AS1 int*)(unsigned long long)(const void*)(gconf + (size_t)r * Cc + c);
      AS3 int* l = (AS3 int*)(unsigned)(unsigned long long)(const void*)&tile[w][r * PITCH + c];
      __builtin_amdgcn_global_load_async_to_lds_b32(g, l, 0, 0);
    }
#else
    for (int idx = lane; idx < nelem; idx += 32) {
      int r = idx / Cc, c = idx - r * Cc;
      tile[w][r * PITCH + c] = gconf[(size_t)r * Cc + c];
    }
#endif
    // -inf pad: cols 81..95 of all 16 rows (disjoint from async destinations)
    for (int idx = lane; idx < 16 * (PITCH - Cc); idx += 32) {
      int r = idx / (PITCH - Cc), c = Cc + (idx - r * (PITCH - Cc));
      tile[w][r * PITCH + c] = NEG_INF;
    }
    // -inf pad: whole rows >= nv (disjoint from async destinations)
    if (nv < 16) {
      for (int idx = lane; idx < (16 - nv) * PITCH; idx += 32)
        tile[w][nv * PITCH + idx] = NEG_INF;
    }
#ifdef HAVE_ASYNC_LDS
#if __has_builtin(__builtin_amdgcn_s_wait_asynccnt)
    __builtin_amdgcn_s_wait_asynccnt(0);
#else
    asm volatile("s_wait_asynccnt 0" ::: "memory");
#endif
#endif
  }
  __builtin_amdgcn_wave_barrier();

  // per-row max: half-lanes scan 48 padded cols each as float4; pads are -inf
  float m = NEG_INF;
  if (nv > 0) {
    const int base = row * PITCH + (lane < 16 ? 0 : 48);
#pragma unroll
    for (int j = 0; j < 12; ++j) {
      const float4 q = *(const float4*)&tile[w][base + 4 * j];
      m = fmaxf(m, fmaxf(fmaxf(q.x, q.y), fmaxf(q.z, q.w)));
    }
  }
  m = fmaxf(m, __shfl_xor(m, 16));

  // WMMA: rowsum[r] = sum_k exp(x[r][k]-max[r]); A=16x32 f16, B=ones, 3 chunks.
  // Branch-free: -inf pads give exp()==0, dead rows give NaN (never read).
  if (nv > 0) {
    v16h ones;
#pragma unroll
    for (int e = 0; e < 16; ++e) ones[e] = (_Float16)1.0f;
    v8f acc = {};
    const int off0 = (lane < 16) ? 0 : 8;
    const int rb = row * PITCH + off0;
#pragma unroll
    for (int ch = 0; ch < 3; ++ch) {
      const int c0 = ch * 32;
      float lo[8], hi[8];
      *(float4*)&lo[0] = *(const float4*)&tile[w][rb + c0];
      *(float4*)&lo[4] = *(const float4*)&tile[w][rb + c0 + 4];
      *(float4*)&hi[0] = *(const float4*)&tile[w][rb + c0 + 16];
      *(float4*)&hi[4] = *(const float4*)&tile[w][rb + c0 + 20];
      v16h a;
#pragma unroll
      for (int e = 0; e < 8; ++e) a[e] = (_Float16)__expf(lo[e] - m);
#pragma unroll
      for (int e = 0; e < 8; ++e) a[8 + e] = (_Float16)__expf(hi[e] - m);
      acc = __builtin_amdgcn_wmma_f32_16x16x32_f16(false, a, false, ones,
                                                   (short)0, acc, false, false);
    }
    // D layout: lane 0 col0 holds rows 0..7 in acc[0..7]; lane 16 rows 8..15
    if (lane == 0) {
#pragma unroll
      for (int i = 0; i < 8; ++i) rowsum[w][i] = acc[i];
    } else if (lane == 16) {
#pragma unroll
      for (int i = 0; i < 8; ++i) rowsum[w][8 + i] = acc[i];
    }
  }
  __syncthreads();

  const bool vl = (nv > 0) && (lane < 16) && (row < nv);
  int lab = 0;
  float sl1 = 0.0f;
  if (vl) {
    const int p = p0 + row;
    const size_t gp = (size_t)b * Pc + p;
    lab = labels[gp];
    const float lse = __logf(rowsum[w][row]) + m;
    const float bg  = lse - tile[w][row * PITCH + 0];
    const float cev = lse - tile[w][row * PITCH + lab];
    cearr[gp] = cev;
    keys[gp]  = float_to_key((lab > 0) ? NEG_INF : bg);
    if (lab > 0) {
#pragma unroll
      for (int i = 0; i < 4; ++i) {
        const float d  = ploc[gp * 4 + i] - gloc[gp * 4 + i];
        const float ad = fabsf(d);
        sl1 += (ad < 1.0f) ? 0.5f * d * d : (ad - 0.5f);
      }
    }
  }
  const unsigned long long bal = __ballot(vl && (lab > 0));
  if (lane == 0 && nv > 0 && bal) atomicAdd(&npr[b], (int)__popcll(bal));
#pragma unroll
  for (int off = 16; off > 0; off >>= 1) sl1 += __shfl_down(sl1, off);
  if (lane == 0) wsl1[w] = (nv > 0) ? sl1 : 0.0f;
  __syncthreads();
  if (tid == 0) {
    float t = 0.0f;
    for (int i = 0; i < WPB; ++i) t += wsl1[i];  // fixed order -> deterministic
    sl1p[blockIdx.y * gridDim.x + blockIdx.x] = t;
  }
}

// ---------------- Phase 2: per-row radix select of num_neg-th largest key -----
__global__ __launch_bounds__(256) void mb_phase2(
    const unsigned* __restrict__ keys, const int* __restrict__ npr,
    unsigned* __restrict__ thr, int* __restrict__ budget)
{
  const int b = blockIdx.x;
  const int tid = threadIdx.x;
  const unsigned* rowk = keys + (size_t)b * Pc;
  __shared__ unsigned hist[256];
  __shared__ unsigned sh_prefix;
  __shared__ int sh_need;
  int k = npr[b] * 3;
  if (k > Pc) k = Pc;
  if (k <= 0) {
    if (tid == 0) { thr[b] = 0xFFFFFFFFu; budget[b] = 0; }
    return;  // k is uniform across block
  }
  if (tid == 0) { sh_prefix = 0u; sh_need = k; }
  __syncthreads();
  for (int shift = 24; shift >= 0; shift -= 8) {
    hist[tid] = 0u;
    __syncthreads();
    const unsigned pfx   = sh_prefix;
    const unsigned hmask = (shift == 24) ? 0u : (0xFFFFFFFFu << (shift + 8));
    for (int i = tid; i < Pc; i += 256) {
      const unsigned key = rowk[i];
      if ((key & hmask) == pfx) atomicAdd(&hist[(key >> shift) & 255u], 1u);
    }
    __syncthreads();
    if (tid == 0) {
      int need = sh_need;
      int d = 255;
      for (; d > 0; --d) {
        const int c = (int)hist[d];
        if (need <= c) break;
        need -= c;
      }
      sh_prefix = pfx | ((unsigned)d << shift);
      sh_need = need;
    }
    __syncthreads();
  }
  if (tid == 0) { thr[b] = sh_prefix; budget[b] = sh_need; }
}

// ---------------- Phase 3: masked CE partial sums ------------------------------
__global__ __launch_bounds__(256) void mb_phase3(
    const unsigned* __restrict__ keys, const float* __restrict__ cearr,
    const int* __restrict__ labels, const unsigned* __restrict__ thr,
    const int* __restrict__ budget, float* __restrict__ objp)
{
  __shared__ float wacc[8];
  const int tid = threadIdx.x;
  const long long idx = (long long)blockIdx.x * 256 + tid;
  float cev = 0.0f;
  if (idx < (long long)Bc * Pc) {
    const int b = (int)(idx / Pc);
    const int p = (int)(idx - (long long)b * Pc);
    const int lab = labels[idx];
    const unsigned key = keys[idx];
    const unsigned t = thr[b];
    const int bud = budget[b];
    bool sel = (lab > 0) || (key > t);
    if (!sel && bud > 0 && key == t) {        // stable-argsort tie break by index
      const unsigned* rowk = keys + (size_t)b * Pc;
      int cnt = 0;
      for (int j = 0; j < p && cnt < bud; ++j) cnt += (rowk[j] == t) ? 1 : 0;
      sel = (cnt < bud);
    }
    if (sel) cev = cearr[idx];
  }
#pragma unroll
  for (int off = 16; off > 0; off >>= 1) cev += __shfl_down(cev, off);
  if ((tid & 31) == 0) wacc[tid >> 5] = cev;
  __syncthreads();
  if (tid == 0) {
    float t = 0.0f;
    for (int i = 0; i < 8; ++i) t += wacc[i];  // fixed order -> deterministic
    objp[blockIdx.x] = t;
  }
}

// ---------------- Phase 4: final deterministic reduction -----------------------
__global__ __launch_bounds__(256) void mb_phase4(
    const float* __restrict__ objp, int nobj,
    const float* __restrict__ sl1p, int nsl1,
    const int* __restrict__ npr, float* __restrict__ out)
{
  __shared__ float so[256], ss[256];
  __shared__ int si[256];
  const int tid = threadIdx.x;
  float o = 0.0f, s = 0.0f;
  int n = 0;
  for (int i = tid; i < nobj; i += 256) o += objp[i];
  for (int i = tid; i < nsl1; i += 256) s += sl1p[i];
  for (int i = tid; i < Bc;   i += 256) n += npr[i];
  so[tid] = o; ss[tid] = s; si[tid] = n;
  __syncthreads();
  for (int off = 128; off > 0; off >>= 1) {
    if (tid < off) {
      so[tid] += so[tid + off];
      ss[tid] += ss[tid + off];
      si[tid] += si[tid + off];
    }
    __syncthreads();
  }
  if (tid == 0) {
    const float np = (float)max(si[0], 1);
    out[0] = so[0] / np;
    out[1] = ss[0] / np;
  }
}

extern "C" void kernel_launch(void* const* d_in, const int* in_sizes, int n_in,
                              void* d_out, int out_size, void* d_ws, size_t ws_size,
                              hipStream_t stream) {
  (void)in_sizes; (void)n_in; (void)out_size; (void)ws_size;
  const float* pred_loc  = (const float*)d_in[0];
  const float* pred_conf = (const float*)d_in[1];
  const float* gt_loc    = (const float*)d_in[2];
  const int*   gt_labels = (const int*)d_in[3];

  const size_t BP = (size_t)Bc * Pc;
  unsigned* keys  = (unsigned*)d_ws;
  float*    cearr = (float*)(keys + BP);
  int*      npr   = (int*)(cearr + BP);
  unsigned* thr   = (unsigned*)(npr + Bc);
  int*      bud   = (int*)(thr + Bc);
  float*    sl1p  = (float*)(bud + Bc);
  float*    objp  = sl1p + (size_t)G1X * Bc;

  (void)hipMemsetAsync(npr, 0, Bc * sizeof(int), stream);

  dim3 g1(G1X, Bc);
  mb_phase1<<<g1, 128, 0, stream>>>(pred_conf, pred_loc, gt_loc, gt_labels,
                                    keys, cearr, npr, sl1p);
  mb_phase2<<<Bc, 256, 0, stream>>>(keys, npr, thr, bud);
  const int nb3 = (int)((BP + 255) / 256);
  mb_phase3<<<nb3, 256, 0, stream>>>(keys, cearr, gt_labels, thr, bud, objp);
  mb_phase4<<<1, 256, 0, stream>>>(objp, nb3, sl1p, G1X * Bc, npr, (float*)d_out);
}
